// AxialAttention_71012989272674
// MI455X (gfx1250) — matrix operand order repaired and checked by reading, not compiled, not run on hardware
//
#include <hip/hip_runtime.h>
#include <hip/hip_bf16.h>

// AxialAttention for MI455X (gfx1250, wave32, WMMA + TDM).
// All matmuls run on v_wmma_f32_16x16x32_f16 (f16 inputs, f32 accumulate).
// ~43 GFLOP total; intermediates (~151MB) stay L2-resident (192MB L2).
// Attention tiles are DMA'd into LDS by the Tensor Data Mover with
// hardware row-padding (bank-conflict avoidance for free), synced via
// s_wait_tensorcnt.

#define EDGE 256
#define NHEAD 8
#define DDIM 32
#define NN 128
#define BB 2
#define QK_SCALE 0.17677669529663687f  // 32^-0.5

typedef __attribute__((ext_vector_type(16))) _Float16 v16h;
typedef __attribute__((ext_vector_type(8)))  _Float16 h8;
typedef __attribute__((ext_vector_type(8)))  float    v8f;
typedef __attribute__((ext_vector_type(4)))  unsigned int u32x4;
typedef __attribute__((ext_vector_type(8)))  int i32x8;
typedef __attribute__((ext_vector_type(4)))  int i32x4;

// ---- WMMA fragment loaders (CDNA5 ISA 7.12.2 layouts) --------------------
// A (16x32 f16, row-major source, stride in halfs):
//   lane l: m = l&15, g = l>>4; halfs t: K = (t/8)*16 + g*8 + (t%8)
__device__ __forceinline__ v16h ldA(const _Float16* base, int stride, int lane) {
  int m = lane & 15, g = lane >> 4;
  const _Float16* r = base + m * stride + g * 8;
  union { v16h v; h8 h[2]; } u;
  u.h[0] = *(const h8*)(r);
  u.h[1] = *(const h8*)(r + 16);
  return u.v;
}
// B (32x16 f16). Source stored transposed: row n holds K contiguously.
//   lane l: n = l&15, g = l>>4; halfs t: K = (t/8)*16 + g*8 + (t%8)
__device__ __forceinline__ v16h ldB(const _Float16* base, int stride, int lane) {
  int n = lane & 15, g = lane >> 4;
  const _Float16* r = base + n * stride + g * 8;
  union { v16h v; h8 h[2]; } u;
  u.h[0] = *(const h8*)(r);
  u.h[1] = *(const h8*)(r + 16);
  return u.v;
}
__device__ __forceinline__ v8f wmma16(v16h a, v16h b, v8f c) {
  return __builtin_amdgcn_wmma_f32_16x16x32_f16(false, a, false, b, (short)0, c,
                                                false, false);
}

// ---- TDM: 1-D tile (n8 x 8B) global -> LDS with optional row padding -----
// D# built per CDNA5 ISA §8.3/8.4. pad_interval v -> pad after 2^(v+1) dwords,
// pad_amount v -> insert (v+1) dwords.
__device__ __forceinline__ void tdm_load_1d(const void* gptr, unsigned lds_addr,
                                            unsigned n8, unsigned pad_interval,
                                            unsigned pad_amount, int pad_en) {
  unsigned long long ga = (unsigned long long)gptr;
  u32x4 g0;
  g0.x = 1u;                                    // count=1 (valid), user mode
  g0.y = lds_addr;                              // LDS byte address
  g0.z = (unsigned)ga;                          // global_addr[31:0]
  g0.w = (unsigned)(ga >> 32) | (2u << 30);     // global_addr[56:32] | type=2
  i32x8 g1;
  g1[0] = (int)((3u << 16) |                    // data_size = 8B
                (pad_en ? (1u << 20) : 0u) |
                (pad_interval << 22) | (pad_amount << 25));
  g1[1] = (int)((n8 & 0xFFFFu) << 16);          // tensor_dim0 lo16
  g1[2] = (int)((n8 >> 16) | (1u << 16));       // tensor_dim0 hi16 | tensor_dim1=1
  g1[3] = (int)(n8 << 16);                      // tile_dim0 (fits 16b)
  g1[4] = 0;                                    // tile_dim1=0, tile_dim2=0
  g1[5] = (int)n8;                              // tensor_dim0_stride lo32
  g1[6] = 0;
  g1[7] = 0;
  i32x4 g2 = {0, 0, 0, 0}, g3 = {0, 0, 0, 0};
#if defined(__clang_major__) && (__clang_major__ >= 23)
  i32x8 g4 = {0, 0, 0, 0, 0, 0, 0, 0};
  __builtin_amdgcn_tensor_load_to_lds(g0, g1, g2, g3, g4, 0);
#else
  __builtin_amdgcn_tensor_load_to_lds(g0, g1, g2, g3, 0);
#endif
}

// dest[b][x][h][y][d]  (Q/K layout: rows of length D=32)
#define IDX4(b,x,h,y,d) ((((((size_t)(b)*NN + (x))*NHEAD + (h))*NN + (y))*DDIM) + (d))
// dest[b][x][h][d][y]  (V-transposed: rows of length N=128)
#define IDXVT(b,x,h,d,y) ((((((size_t)(b)*NN + (x))*NHEAD + (h))*DDIM + (d))*NN) + (y))

// ---- Kernel 1: LayerNorm fp32 -> f16 -------------------------------------
__global__ __launch_bounds__(256) void ln_kernel(const float* __restrict__ e,
                                                 const float* __restrict__ lw,
                                                 const float* __restrict__ lb,
                                                 _Float16* __restrict__ eln) {
  int lane = threadIdx.x & 31, w = threadIdx.x >> 5;
  size_t row = (size_t)blockIdx.x * 8 + w;
  const float* src = e + row * EDGE;
  float v[8];
  float4 a0 = *(const float4*)(src + lane * 8);
  float4 a1 = *(const float4*)(src + lane * 8 + 4);
  v[0]=a0.x; v[1]=a0.y; v[2]=a0.z; v[3]=a0.w;
  v[4]=a1.x; v[5]=a1.y; v[6]=a1.z; v[7]=a1.w;
  float s = 0.f, s2 = 0.f;
#pragma unroll
  for (int t = 0; t < 8; ++t) { s += v[t]; s2 += v[t]*v[t]; }
#pragma unroll
  for (int o = 16; o; o >>= 1) { s += __shfl_xor(s, o, 32); s2 += __shfl_xor(s2, o, 32); }
  float mu = s * (1.f/EDGE);
  float var = s2 * (1.f/EDGE) - mu * mu;
  float rs = rsqrtf(var + 1e-5f);
  union { h8 h; uint4 u; } out;
#pragma unroll
  for (int t = 0; t < 8; ++t) {
    int c = lane * 8 + t;
    out.h[t] = (_Float16)((v[t] - mu) * rs * lw[c] + lb[c]);
  }
  *(uint4*)(eln + row * EDGE + lane * 8) = out.u;
}

// ---- Kernel 2: QKV projection GEMM + layout scatter -----------------------
// C[32768x768] = eln[32768x256] @ W[256x768] + bias, then scatter into
// Q[b][j][h][i][d]*SCALE, K[b][x][h][y][d], Vt[b][x][h][d][y].
#define AS_S 72   // padded LDS strides (halfs): 144B rows -> 4-way max conflicts
__global__ __launch_bounds__(256) void qkv_gemm_kernel(
    const _Float16* __restrict__ eln, const float* __restrict__ W,
    const float* __restrict__ bias, _Float16* __restrict__ Qb,
    _Float16* __restrict__ Kb, _Float16* __restrict__ Vtb, int isIn) {
  __shared__ _Float16 As[128 * AS_S];
  __shared__ _Float16 Ws[128 * AS_S];
  int tid = threadIdx.x, lane = tid & 31, w = tid >> 5;
  int Mt = blockIdx.x, Nt = blockIdx.y;
  v8f acc[8] = {};
  for (int kc4 = 0; kc4 < 4; ++kc4) {   // K = 256 in 4 chunks of 64
#pragma unroll
    for (int it = 0; it < 4; ++it) {    // A tile 128x64 f16 (coalesced 16B)
      int lin = it * 256 + tid, r = lin >> 3, c = lin & 7;
      *(uint4*)(As + r * AS_S + c * 8) =
          *(const uint4*)(eln + ((size_t)(Mt * 128 + r)) * EDGE + kc4 * 64 + c * 8);
    }
#pragma unroll
    for (int it = 0; it < 32; ++it) {   // W^T tile: Ws[n][kk] = W[kk][Nt*128+n]
      int lin = it * 256 + tid, kk = lin >> 7, n = lin & 127;
      Ws[n * AS_S + kk] = (_Float16)W[(size_t)(kc4 * 64 + kk) * 768 + Nt * 128 + n];
    }
    __syncthreads();
#pragma unroll
    for (int kc = 0; kc < 2; ++kc) {
      v16h a = ldA(As + (w * 16) * AS_S + kc * 32, AS_S, lane);
      v16h bf[8];
#pragma unroll
      for (int nt = 0; nt < 8; ++nt)    // preload all B-frags: overlapped waits
        bf[nt] = ldB(Ws + (nt * 16) * AS_S + kc * 32, AS_S, lane);
#pragma unroll
      for (int nt = 0; nt < 8; ++nt)    // 8 independent accumulators
        acc[nt] = wmma16(a, bf[nt], acc[nt]);
    }
    __syncthreads();
  }
  int g = lane >> 4, nl = lane & 15;
#pragma unroll
  for (int nt = 0; nt < 8; ++nt) {
#pragma unroll
    for (int p = 0; p < 8; ++p) {
      int Rg = Mt * 128 + w * 16 + p + 8 * g;
      int Cg = Nt * 128 + nt * 16 + nl;
      float vv = acc[nt][p] + bias[Cg];
      int b = Rg >> 14, i1 = (Rg >> 7) & 127, j1 = Rg & 127;
      int sec = Cg >> 8, cc = Cg & 255, d = cc >> 3, h = cc & 7;
      if (sec == 0) {                              // Q: needed over i at fixed j
        Qb[IDX4(b, j1, h, i1, d)] = (_Float16)(vv * QK_SCALE);
      } else if (sec == 1) {                       // K
        int x = isIn ? i1 : j1, y = isIn ? j1 : i1;
        Kb[IDX4(b, x, h, y, d)] = (_Float16)vv;
      } else {                                     // V (transposed)
        int x = isIn ? i1 : j1, y = isIn ? j1 : i1;
        Vtb[IDXVT(b, x, h, d, y)] = (_Float16)vv;
      }
    }
  }
}

// ---- Kernel 3: attention per (b, j, h) ------------------------------------
#define QK_S 40    // 80B rows  (64B data + 16B TDM pad)
#define VT_S 136   // 272B rows (256B data + 16B TDM pad)
__global__ __launch_bounds__(128) void attn_kernel(
    const _Float16* __restrict__ Qb, const _Float16* __restrict__ Kb,
    const _Float16* __restrict__ Vtb, const float* __restrict__ mask,
    _Float16* __restrict__ Va, int isIn) {
  __shared__ _Float16 Qs[128 * QK_S];
  __shared__ _Float16 Ks[128 * QK_S];
  __shared__ _Float16 Vts[32 * VT_S];
  __shared__ _Float16 Ps[128 * VT_S];
  int tid = threadIdx.x, lane = tid & 31, w = tid >> 5;
  int g = lane >> 4, nl = lane & 15;
  int bid = blockIdx.x;
  int h = bid & 7, j = (bid >> 3) & 127, b = bid >> 10;
  size_t tile = (((size_t)b * NN + j) * NHEAD + h) * (NN * DDIM); // 4096 halfs

  // TDM: each of waves 0..2 DMA's one 8KB tile into LDS, with hardware
  // padding re-striding rows (Q/K: 64B rows + 16B pad; Vt: 256B + 16B pad).
  if (w == 0) {
    tdm_load_1d(Qb + tile, (unsigned)(uintptr_t)Qs, 1024, 3, 3, 1);
  } else if (w == 1) {
    tdm_load_1d(Kb + tile, (unsigned)(uintptr_t)Ks, 1024, 3, 3, 1);
  } else if (w == 2) {
    tdm_load_1d(Vtb + tile, (unsigned)(uintptr_t)Vts, 1024, 5, 3, 1);
  }
  __builtin_amdgcn_s_wait_tensorcnt(0);
  __syncthreads();

#pragma unroll
  for (int rti = 0; rti < 2; ++rti) {
    int rt = 2 * w + rti;                 // 16-row strip owned by this wave
    v8f acc[8] = {};
    v16h a = ldA(Qs + rt * 16 * QK_S, QK_S, lane);
    v16h kf[8];
#pragma unroll
    for (int ct = 0; ct < 8; ++ct)        // preload all K B-frags
      kf[ct] = ldB(Ks + ct * 16 * QK_S, QK_S, lane);
#pragma unroll
    for (int ct = 0; ct < 8; ++ct)        // logits: single WMMA per tile (D=32)
      acc[ct] = wmma16(a, kf[ct], acc[ct]);
    float rm[8], rs[8], inv[8];
#pragma unroll
    for (int p = 0; p < 8; ++p) { rm[p] = -3.0e38f; rs[p] = 0.f; }
#pragma unroll
    for (int ct = 0; ct < 8; ++ct) {
#pragma unroll
      for (int p = 0; p < 8; ++p) {
        int ri = rt * 16 + p + 8 * g, col = ct * 16 + nl;
        int mi = isIn ? ri : col, mk = isIn ? col : ri;
        float t = acc[ct][p] +
                  mask[(((size_t)b * NN + mi) * NN + mk) * NHEAD + h];
        acc[ct][p] = t;
        rm[p] = fmaxf(rm[p], t);
      }
    }
#pragma unroll
    for (int p = 0; p < 8; ++p)           // rows live in 16-lane halves
#pragma unroll
      for (int o = 8; o; o >>= 1) rm[p] = fmaxf(rm[p], __shfl_xor(rm[p], o, 16));
#pragma unroll
    for (int ct = 0; ct < 8; ++ct)
#pragma unroll
      for (int p = 0; p < 8; ++p) {
        float ex = __expf(acc[ct][p] - rm[p]);
        acc[ct][p] = ex;
        rs[p] += ex;
      }
#pragma unroll
    for (int p = 0; p < 8; ++p) {
#pragma unroll
      for (int o = 8; o; o >>= 1) rs[p] += __shfl_xor(rs[p], o, 16);
      inv[p] = 1.f / rs[p];
    }
#pragma unroll
    for (int ct = 0; ct < 8; ++ct)        // probs -> LDS (intra-wave reuse only)
#pragma unroll
      for (int p = 0; p < 8; ++p)
        Ps[(rt * 16 + p + 8 * g) * VT_S + ct * 16 + nl] = (_Float16)acc[ct][p];
#pragma unroll
    for (int dn = 0; dn < 2; ++dn) {      // O = P @ V : K=128 chained WMMAs
      v16h pa[4], vb[4];
#pragma unroll
      for (int kc = 0; kc < 4; ++kc) {
        pa[kc] = ldA(Ps + rt * 16 * VT_S + kc * 32, VT_S, lane);
        vb[kc] = ldB(Vts + dn * 16 * VT_S + kc * 32, VT_S, lane);
      }
      v8f o = {};
#pragma unroll
      for (int kc = 0; kc < 4; ++kc)
        o = wmma16(pa[kc], vb[kc], o);
      int off = isIn ? 0 : EDGE;
#pragma unroll
      for (int p = 0; p < 8; ++p) {
        int irow = rt * 16 + p + 8 * g, d = dn * 16 + nl;
        Va[(((size_t)b * NN + irow) * NN + j) * (2 * EDGE) + off + d * 8 + h] =
            (_Float16)(o[p] * inv[p]);
      }
    }
  }
}

// ---- Kernel 4: output projection [32768,512]@[512,256] + bias -> fp32 -----
__global__ __launch_bounds__(256) void out_gemm_kernel(
    const _Float16* __restrict__ Va, const float* __restrict__ Wo,
    const float* __restrict__ bo, float* __restrict__ out) {
  __shared__ _Float16 As[128 * AS_S];
  __shared__ _Float16 Ws[128 * AS_S];
  int tid = threadIdx.x, lane = tid & 31, w = tid >> 5;
  int Mt = blockIdx.x, Nt = blockIdx.y;
  v8f acc[8] = {};
  for (int kc4 = 0; kc4 < 8; ++kc4) {    // K = 512 in 8 chunks of 64
#pragma unroll
    for (int it = 0; it < 4; ++it) {
      int lin = it * 256 + tid, r = lin >> 3, c = lin & 7;
      *(uint4*)(As + r * AS_S + c * 8) =
          *(const uint4*)(Va + ((size_t)(Mt * 128 + r)) * 512 + kc4 * 64 + c * 8);
    }
#pragma unroll
    for (int it = 0; it < 32; ++it) {
      int lin = it * 256 + tid, kk = lin >> 7, n = lin & 127;
      Ws[n * AS_S + kk] = (_Float16)Wo[(size_t)(kc4 * 64 + kk) * 256 + Nt * 128 + n];
    }
    __syncthreads();
#pragma unroll
    for (int kc = 0; kc < 2; ++kc) {
      v16h a = ldA(As + (w * 16) * AS_S + kc * 32, AS_S, lane);
      v16h bf[8];
#pragma unroll
      for (int nt = 0; nt < 8; ++nt)
        bf[nt] = ldB(Ws + (nt * 16) * AS_S + kc * 32, AS_S, lane);
#pragma unroll
      for (int nt = 0; nt < 8; ++nt)
        acc[nt] = wmma16(a, bf[nt], acc[nt]);
    }
    __syncthreads();
  }
  int g = lane >> 4, nl = lane & 15;
#pragma unroll
  for (int nt = 0; nt < 8; ++nt)
#pragma unroll
    for (int p = 0; p < 8; ++p) {
      int Rg = Mt * 128 + w * 16 + p + 8 * g;
      int Cg = Nt * 128 + nt * 16 + nl;
      out[(size_t)Rg * EDGE + Cg] = acc[nt][p] + bo[Cg];
    }
}

extern "C" void kernel_launch(void* const* d_in, const int* in_sizes, int n_in,
                              void* d_out, int out_size, void* d_ws, size_t ws_size,
                              hipStream_t stream) {
  const float* e        = (const float*)d_in[0];
  const float* mask     = (const float*)d_in[1];
  const float* ln_w     = (const float*)d_in[2];
  const float* ln_b     = (const float*)d_in[3];
  const float* Wqkv_in  = (const float*)d_in[4];
  const float* bqkv_in  = (const float*)d_in[5];
  const float* Wqkv_out = (const float*)d_in[6];
  const float* bqkv_out = (const float*)d_in[7];
  const float* Wo       = (const float*)d_in[8];
  const float* bo       = (const float*)d_in[9];
  char* ws = (char*)d_ws;
  // Workspace map (needs ~151MB): each attention tensor is 16.78MB f16.
  const size_t T = (size_t)BB * NN * NN * EDGE * sizeof(_Float16);  // 16,777,216
  _Float16* eln   = (_Float16*)(ws + 0 * T);
  _Float16* Qin   = (_Float16*)(ws + 1 * T);
  _Float16* Kin   = (_Float16*)(ws + 2 * T);
  _Float16* Vtin  = (_Float16*)(ws + 3 * T);
  _Float16* Qout  = (_Float16*)(ws + 4 * T);
  _Float16* Kout  = (_Float16*)(ws + 5 * T);
  _Float16* Vtout = (_Float16*)(ws + 6 * T);
  _Float16* Va    = (_Float16*)(ws + 7 * T);  // 33.5MB ([32768][512] f16)

  ln_kernel<<<4096, 256, 0, stream>>>(e, ln_w, ln_b, eln);
  qkv_gemm_kernel<<<dim3(256, 6), 256, 0, stream>>>(eln, Wqkv_in, bqkv_in,
                                                    Qin, Kin, Vtin, 1);
  qkv_gemm_kernel<<<dim3(256, 6), 256, 0, stream>>>(eln, Wqkv_out, bqkv_out,
                                                    Qout, Kout, Vtout, 0);
  attn_kernel<<<BB * NN * NHEAD, 128, 0, stream>>>(Qin, Kin, Vtin, mask, Va, 1);
  attn_kernel<<<BB * NN * NHEAD, 128, 0, stream>>>(Qout, Kout, Vtout, mask, Va, 0);
  out_gemm_kernel<<<dim3(256, 2), 256, 0, stream>>>(Va, Wo, bo, (float*)d_out);
}
